// MLAEncoderBlock_3667902071067
// MI455X (gfx1250) — matrix operand order repaired and checked by reading, not compiled, not run on hardware
//
#include <hip/hip_runtime.h>
#include <hip/hip_bf16.h>
#include <cstdint>

typedef __attribute__((ext_vector_type(16))) _Float16 v16h;
typedef __attribute__((ext_vector_type(8)))  _Float16 v8h;
typedef __attribute__((ext_vector_type(4)))  _Float16 v4h;
typedef __attribute__((ext_vector_type(8)))  float    v8f;

#define DEVINL __device__ __forceinline__

// ---------------------------------------------------------------------------
// WMMA helper (CDNA5 16x16x32 f16 -> f32 accumulate)
// ---------------------------------------------------------------------------
DEVINL v8f wmma32(v16h a, v16h b, v8f c) {
  return __builtin_amdgcn_wmma_f32_16x16x32_f16(
      /*neg_a=*/false, a, /*neg_b=*/false, b,
      /*c_mod=*/(short)0, c, /*reuse_a=*/false, /*reuse_b=*/false);
}

// A fragment: 16 rows x 32 K, row-major in LDS (row stride ldm halfs).
// lanes 0-15: row = lane, K 0..7 & 16..23 ; lanes 16-31: row = lane-16, K 8..15 & 24..31
DEVINL v16h ldsA(const _Float16* base, int ldm, int lane) {
  const _Float16* p = base + (lane & 15) * ldm + ((lane & 16) ? 8 : 0);
  v8h lo = *(const v8h*)p;
  v8h hi = *(const v8h*)(p + 16);
  v16h r;
#pragma unroll
  for (int i = 0; i < 8; ++i) { r[i] = lo[i]; r[8 + i] = hi[i]; }
  return r;
}

// B fragment: 32 K-rows x 16 cols, row-major in LDS (row stride ldn halfs).
// lanes 0-15: col = lane, K 0..7 & 16..23 ; lanes 16-31: col = lane-16, K 8..15 & 24..31
DEVINL v16h ldsB(const _Float16* base, int ldn, int lane) {
  const _Float16* p = base + (lane & 15) + ((lane & 16) ? 8 : 0) * ldn;
  v16h r;
#pragma unroll
  for (int i = 0; i < 8; ++i) { r[i] = p[i * ldn]; r[8 + i] = p[(16 + i) * ldn]; }
  return r;
}

// ---------------------------------------------------------------------------
// f32 -> f16 cast (vectorized x4)
// ---------------------------------------------------------------------------
__global__ __launch_bounds__(256) void cast_kernel(const float* __restrict__ s,
                                                   _Float16* __restrict__ d, int n4) {
  int i = blockIdx.x * 256 + threadIdx.x;
  if (i < n4) {
    float4 v = ((const float4*)s)[i];
    v4h o;
    o[0] = (_Float16)v.x; o[1] = (_Float16)v.y;
    o[2] = (_Float16)v.z; o[3] = (_Float16)v.w;
    ((v4h*)d)[i] = o;
  }
}

// ---------------------------------------------------------------------------
// Tiled WMMA GEMM, LDS ping-pong double buffered.
// C[M,N] = act(A[M,K] * W[K,N] + bias[N])
// block tile 128x128, K-step 32, 8 waves (4 M x 2 N), wave tile 32x64.
// ---------------------------------------------------------------------------
template <bool GELU_ACT, typename OutT>
__global__ __launch_bounds__(256) void gemm_kernel(const _Float16* __restrict__ A,
                                                   const _Float16* __restrict__ W,
                                                   const float* __restrict__ bias,
                                                   OutT* __restrict__ C,
                                                   int M, int K, int N) {
  constexpr int TM = 128, TN = 128, TK = 32;
  constexpr int LDA = TK + 8;   // 80B row stride
  constexpr int LDB = TN + 8;   // 272B row stride
  __shared__ alignas(16) _Float16 As[2][TM * LDA];
  __shared__ alignas(16) _Float16 Bs[2][TK * LDB];

  const int tid  = threadIdx.x;
  const int lane = tid & 31;
  const int wid  = tid >> 5;
  const int waveM = wid & 3;   // 0..3
  const int waveN = wid >> 2;  // 0..1
  const int rowBase = blockIdx.y * TM;
  const int colBase = blockIdx.x * TN;

  // staging assignments
  const int rA = tid >> 1, cA = tid & 1;   // A: 128 rows x (2 x 16h)
  const int rB = tid >> 3, cB = tid & 7;   // B: 32 rows x (8 x 16h)
  const _Float16* gA = A + (size_t)(rowBase + rA) * K + cA * 16;
  const _Float16* gB = W + (size_t)rB * N + colBase + cB * 16;
  _Float16* const sA = (_Float16*)As[0] + rA * LDA + cA * 16;  // buffer 0 slot
  _Float16* const sB = (_Float16*)Bs[0] + rB * LDB + cB * 16;
  const int bufStrideA = TM * LDA, bufStrideB = TK * LDB;

  v8f acc[2][4];
#pragma unroll
  for (int i = 0; i < 2; ++i)
#pragma unroll
    for (int j = 0; j < 4; ++j)
#pragma unroll
      for (int r = 0; r < 8; ++r) acc[i][j][r] = 0.0f;

  // preload tile 0 into buffer 0
  {
    v8h a0 = *(const v8h*)gA, a1 = *(const v8h*)(gA + 8);
    v8h b0 = *(const v8h*)gB, b1 = *(const v8h*)(gB + 8);
    *(v8h*)sA = a0; *(v8h*)(sA + 8) = a1;
    *(v8h*)sB = b0; *(v8h*)(sB + 8) = b1;
  }
  __syncthreads();

  const int nk = K / TK;
  for (int it = 0; it < nk; ++it) {
    const int cur = it & 1, nxt = cur ^ 1;
    v8h a0 = {}, a1 = {}, b0 = {}, b1 = {};
    const bool has = (it + 1 < nk);
    if (has) {  // issue next tile's global loads (hidden behind WMMA)
      const _Float16* pa = gA + (size_t)(it + 1) * TK;
      const _Float16* pb = gB + (size_t)(it + 1) * TK * N;
      a0 = *(const v8h*)pa; a1 = *(const v8h*)(pa + 8);
      b0 = *(const v8h*)pb; b1 = *(const v8h*)(pb + 8);
    }

    v16h af[2], bf[4];
#pragma unroll
    for (int i = 0; i < 2; ++i)
      af[i] = ldsA(As[cur] + (waveM * 32 + i * 16) * LDA, LDA, lane);
#pragma unroll
    for (int j = 0; j < 4; ++j)
      bf[j] = ldsB(Bs[cur] + waveN * 64 + j * 16, LDB, lane);
#pragma unroll
    for (int i = 0; i < 2; ++i)
#pragma unroll
      for (int j = 0; j < 4; ++j)
        acc[i][j] = wmma32(af[i], bf[j], acc[i][j]);

    if (has) {  // commit staged registers into the other buffer
      _Float16* da = sA + nxt * bufStrideA;
      _Float16* db = sB + nxt * bufStrideB;
      *(v8h*)da = a0; *(v8h*)(da + 8) = a1;
      *(v8h*)db = b0; *(v8h*)(db + 8) = b1;
    }
    __syncthreads();
  }

  // epilogue: bias (+ exact GELU), store f32 or f16
  const int mlo = (lane & 16) ? 8 : 0;
  const int cn  = lane & 15;
#pragma unroll
  for (int i = 0; i < 2; ++i) {
#pragma unroll
    for (int j = 0; j < 4; ++j) {
      int col = colBase + waveN * 64 + j * 16 + cn;
      float bv = bias[col];
#pragma unroll
      for (int r = 0; r < 8; ++r) {
        int row = rowBase + waveM * 32 + i * 16 + mlo + r;
        float v = acc[i][j][r] + bv;
        if (GELU_ACT) v = 0.5f * v * (1.0f + erff(v * 0.70710678118f));
        C[(size_t)row * N + col] = (OutT)v;
      }
    }
  }
}

// ---------------------------------------------------------------------------
// Flash attention: one block = 4 waves = 64 query rows for one (b,h).
// Computes S^T = K * Q^T so K/V stage with vectorized b128 stores every tile;
// only Q is transposed (once per block). Online softmax, WMMA P*V.
// ---------------------------------------------------------------------------
__global__ __launch_bounds__(128) void attn_kernel(const _Float16* __restrict__ Qg,
                                                   const _Float16* __restrict__ Kg,
                                                   const _Float16* __restrict__ Vg,
                                                   const float* __restrict__ biasg,
                                                   _Float16* __restrict__ Og) {
  constexpr int Nseq = 2048, E = 1024;
  constexpr int QT_LD = 72;  // Qt: [64 d][64 q + pad]
  constexpr int KS_LD = 72;  // Ks: [32 keys][64 d + pad]
  constexpr int V_LD  = 72;  // Vs: [32 keys][64 d + pad]
  constexpr int P_LD  = 40;  // Ps: [16 q][32 keys + pad]

  __shared__ alignas(16) _Float16 Qt[64 * QT_LD];
  __shared__ alignas(16) _Float16 Ks[32 * KS_LD];
  __shared__ alignas(16) _Float16 Vs[32 * V_LD];
  __shared__ alignas(16) float    Ss[4][16 * 32];
  __shared__ alignas(16) _Float16 Ps[4][16 * P_LD];
  __shared__ float mRow[4][16], lRow[4][16], fRow[4][16];

  const int bh = blockIdx.x;          // 0..31
  const int b = bh >> 4, h = bh & 15;
  const int qt = blockIdx.y;          // 0..31 (64 q rows each)
  const int tid = threadIdx.x;
  const int lane = tid & 31;
  const int w = tid >> 5;             // wave id (0..3)
  const size_t headOff = (size_t)(b * Nseq) * E + (size_t)h * 64;

  {  // load Q tile 64x64 and transpose into Qt[d][q] (once per block)
    int r = tid >> 1, c = tid & 1;   // r: q row 0..63, c: d half 0..1
    const _Float16* src = Qg + headOff + (size_t)(qt * 64 + r) * E + c * 32;
    v8h u[4];
#pragma unroll
    for (int u0 = 0; u0 < 4; ++u0) u[u0] = *(const v8h*)(src + u0 * 8);
#pragma unroll
    for (int d0 = 0; d0 < 32; ++d0)
      Qt[(c * 32 + d0) * QT_LD + r] = u[d0 >> 3][d0 & 7];
  }
  if (tid < 64) {
    int ww = tid >> 4, r = tid & 15;
    mRow[ww][r] = -1e30f;
    lRow[ww][r] = 0.0f;
  }

  v8f O[4];
#pragma unroll
  for (int j = 0; j < 4; ++j)
#pragma unroll
    for (int r = 0; r < 8; ++r) O[j][r] = 0.0f;

  __syncthreads();

  const int mlo = (lane & 16) ? 8 : 0;
  const int cn  = lane & 15;

  for (int kt = 0; kt < Nseq / 32; ++kt) {
    {  // stage K and V tiles row-major: 32 keys x 64 d, vectorized
      int r = tid >> 2, c = tid & 3;  // r: key 0..31, c: 16-half chunk
      const _Float16* ks = Kg + headOff + (size_t)(kt * 32 + r) * E + c * 16;
      const _Float16* vs = Vg + headOff + (size_t)(kt * 32 + r) * E + c * 16;
      v8h k0 = *(const v8h*)ks, k1 = *(const v8h*)(ks + 8);
      v8h v0 = *(const v8h*)vs, v1 = *(const v8h*)(vs + 8);
      _Float16* kd = Ks + r * KS_LD + c * 16;
      _Float16* vd = Vs + r * V_LD + c * 16;
      *(v8h*)kd = k0; *(v8h*)(kd + 8) = k1;
      *(v8h*)vd = v0; *(v8h*)(vd + 8) = v1;
    }
    __syncthreads();

    // S^T tile: A = K rows (keys), B = Qt cols (this wave's 16 q rows)
    v8f st[2];
#pragma unroll
    for (int j = 0; j < 2; ++j)
#pragma unroll
      for (int r = 0; r < 8; ++r) st[j][r] = 0.0f;
#pragma unroll
    for (int dstep = 0; dstep < 2; ++dstep) {
      v16h bq = ldsB(Qt + dstep * 32 * QT_LD + w * 16, QT_LD, lane);
#pragma unroll
      for (int ktile = 0; ktile < 2; ++ktile) {
        v16h ak = ldsA(Ks + ktile * 16 * KS_LD + dstep * 32, KS_LD, lane);
        st[ktile] = wmma32(ak, bq, st[ktile]);
      }
    }
    // scatter S^T into Ss[q][key] (raw scores; scale+bias in softmax pass)
#pragma unroll
    for (int ktile = 0; ktile < 2; ++ktile) {
#pragma unroll
      for (int r = 0; r < 8; ++r) {
        int key = ktile * 16 + mlo + r;
        Ss[w][cn * 32 + key] = st[ktile][r];
      }
    }
    __syncthreads();

    // online softmax: 2 lanes per q row, 16 keys each (coalesced bias read)
    {
      int row = lane >> 1, seg = lane & 1;
      int qrow = qt * 64 + w * 16 + row;
      const float* sp = &Ss[w][row * 32 + seg * 16];
      const float* bp = biasg + (size_t)qrow * Nseq + kt * 32 + seg * 16;
      float vbuf[16];
      float tm = -1e30f;
#pragma unroll
      for (int i = 0; i < 16; ++i) {
        vbuf[i] = sp[i] * 0.125f + bp[i];
        tm = fmaxf(tm, vbuf[i]);
      }
      tm = fmaxf(tm, __shfl_xor(tm, 1));
      float mo = mRow[w][row];
      float mn = fmaxf(mo, tm);
      float sum = 0.0f;
      _Float16* pp = &Ps[w][row * P_LD + seg * 16];
#pragma unroll
      for (int i = 0; i < 16; ++i) {
        float p = __expf(vbuf[i] - mn);
        sum += p;
        pp[i] = (_Float16)p;
      }
      sum += __shfl_xor(sum, 1);
      float fac = __expf(mo - mn);
      if (seg == 0) {
        mRow[w][row] = mn;
        lRow[w][row] = lRow[w][row] * fac + sum;
        fRow[w][row] = fac;
      }
    }
    __syncthreads();

    // rescale O and accumulate P*V
    {
      v8f fv;
#pragma unroll
      for (int r = 0; r < 8; ++r) fv[r] = fRow[w][mlo + r];
#pragma unroll
      for (int j = 0; j < 4; ++j) O[j] *= fv;
      v16h ap = ldsA(Ps[w], P_LD, lane);
#pragma unroll
      for (int j = 0; j < 4; ++j) {
        v16h bv = ldsB(Vs + j * 16, V_LD, lane);
        O[j] = wmma32(ap, bv, O[j]);
      }
    }
    __syncthreads();
  }

  // finalize: divide by l, write f16 at [b, row, h*64 + d]
  {
    v8f inv;
#pragma unroll
    for (int r = 0; r < 8; ++r) inv[r] = 1.0f / lRow[w][mlo + r];
#pragma unroll
    for (int j = 0; j < 4; ++j) {
#pragma unroll
      for (int r = 0; r < 8; ++r) {
        int row = qt * 64 + w * 16 + mlo + r;
        Og[headOff + (size_t)row * E + j * 16 + cn] = (_Float16)(O[j][r] * inv[r]);
      }
    }
  }
}

// ---------------------------------------------------------------------------
// Fused residual + LayerNorm (E=1024), one block per row.
// ---------------------------------------------------------------------------
template <bool WF32, bool WF16>
__global__ __launch_bounds__(256) void ln_kernel(const float* __restrict__ X,
                                                 const float* __restrict__ R,
                                                 const float* __restrict__ wv,
                                                 const float* __restrict__ bv,
                                                 float* __restrict__ o32,
                                                 _Float16* __restrict__ o16) {
  constexpr int E = 1024;
  __shared__ float sred[16];
  const int row = blockIdx.x, tid = threadIdx.x;
  const float4 xv = ((const float4*)(X + (size_t)row * E))[tid];
  const float4 rv = ((const float4*)(R + (size_t)row * E))[tid];
  float v[4] = {xv.x + rv.x, xv.y + rv.y, xv.z + rv.z, xv.w + rv.w};
  float s = v[0] + v[1] + v[2] + v[3];
  float q = v[0] * v[0] + v[1] * v[1] + v[2] * v[2] + v[3] * v[3];
#pragma unroll
  for (int off = 16; off > 0; off >>= 1) {
    s += __shfl_xor(s, off);
    q += __shfl_xor(q, off);
  }
  int lane = tid & 31, wid = tid >> 5;
  if (lane == 0) { sred[wid] = s; sred[8 + wid] = q; }
  __syncthreads();
  if (tid == 0) {
    float ts = 0.0f, tq = 0.0f;
#pragma unroll
    for (int i = 0; i < 8; ++i) { ts += sred[i]; tq += sred[8 + i]; }
    sred[0] = ts;
    sred[8] = tq;
  }
  __syncthreads();
  const float mu = sred[0] * (1.0f / E);
  const float var = sred[8] * (1.0f / E) - mu * mu;
  const float rstd = rsqrtf(var + 1e-5f);
  const int c = tid * 4;
  const float4 wq = *(const float4*)(wv + c);
  const float4 bq = *(const float4*)(bv + c);
  float y[4];
  y[0] = (v[0] - mu) * rstd * wq.x + bq.x;
  y[1] = (v[1] - mu) * rstd * wq.y + bq.y;
  y[2] = (v[2] - mu) * rstd * wq.z + bq.z;
  y[3] = (v[3] - mu) * rstd * wq.w + bq.w;
  if (WF32) {
    float4 o = {y[0], y[1], y[2], y[3]};
    ((float4*)(o32 + (size_t)row * E))[tid] = o;
  }
  if (WF16) {
    v4h o;
    o[0] = (_Float16)y[0]; o[1] = (_Float16)y[1];
    o[2] = (_Float16)y[2]; o[3] = (_Float16)y[3];
    ((v4h*)(o16 + (size_t)row * E))[tid] = o;
  }
}

// ---------------------------------------------------------------------------
// host-side orchestration
// ---------------------------------------------------------------------------
extern "C" void kernel_launch(void* const* d_in, const int* in_sizes, int n_in,
                              void* d_out, int out_size, void* d_ws, size_t ws_size,
                              hipStream_t stream) {
  constexpr int EMB = 1024, LAT = 256, FFN = 4096, Nseq = 2048, Bb = 2;
  constexpr int M = Bb * Nseq;  // 4096 rows

  const float* x      = (const float*)d_in[0];
  const float* abias  = (const float*)d_in[1];
  const float* Wq = (const float*)d_in[2];  const float* bq  = (const float*)d_in[3];
  const float* WaK = (const float*)d_in[4]; const float* baK = (const float*)d_in[5];
  const float* WbK = (const float*)d_in[6]; const float* bbK = (const float*)d_in[7];
  const float* WaV = (const float*)d_in[8]; const float* baV = (const float*)d_in[9];
  const float* WbV = (const float*)d_in[10]; const float* bbV = (const float*)d_in[11];
  const float* Wo = (const float*)d_in[12]; const float* bo = (const float*)d_in[13];
  const float* W1 = (const float*)d_in[14]; const float* b1 = (const float*)d_in[15];
  const float* W2 = (const float*)d_in[16]; const float* b2 = (const float*)d_in[17];
  const float* ln1w = (const float*)d_in[18]; const float* ln1b = (const float*)d_in[19];
  const float* ln2w = (const float*)d_in[20]; const float* ln2b = (const float*)d_in[21];

  char* ws = (char*)d_ws;
  size_t off = 0;
  auto alloc = [&](size_t bytes) -> void* {
    void* p = ws + off;
    off += (bytes + 255) & ~(size_t)255;
    return p;
  };

  _Float16* x_h    = (_Float16*)alloc((size_t)M * EMB * 2);
  _Float16* Wq_h   = (_Float16*)alloc((size_t)EMB * EMB * 2);
  _Float16* WaK_h  = (_Float16*)alloc((size_t)EMB * LAT * 2);
  _Float16* WbK_h  = (_Float16*)alloc((size_t)LAT * EMB * 2);
  _Float16* WaV_h  = (_Float16*)alloc((size_t)EMB * LAT * 2);
  _Float16* WbV_h  = (_Float16*)alloc((size_t)LAT * EMB * 2);
  _Float16* Wo_h   = (_Float16*)alloc((size_t)EMB * EMB * 2);
  _Float16* W1_h   = (_Float16*)alloc((size_t)EMB * FFN * 2);
  _Float16* W2_h   = (_Float16*)alloc((size_t)FFN * EMB * 2);
  _Float16* q_h    = (_Float16*)alloc((size_t)M * EMB * 2);
  _Float16* ka_h   = (_Float16*)alloc((size_t)M * LAT * 2);
  _Float16* k_h    = (_Float16*)alloc((size_t)M * EMB * 2);
  _Float16* va_h   = (_Float16*)alloc((size_t)M * LAT * 2);
  _Float16* v_h    = (_Float16*)alloc((size_t)M * EMB * 2);
  _Float16* attn_h = (_Float16*)alloc((size_t)M * EMB * 2);
  float*    attnO  = (float*)alloc((size_t)M * EMB * 4);
  float*    h32    = (float*)alloc((size_t)M * EMB * 4);
  _Float16* h16    = (_Float16*)alloc((size_t)M * EMB * 2);
  _Float16* ffn1_h = (_Float16*)alloc((size_t)M * FFN * 2);
  float*    ffn2   = (float*)alloc((size_t)M * EMB * 4);

  auto cast = [&](const float* s, _Float16* d, size_t n) {
    int n4 = (int)(n / 4);
    cast_kernel<<<dim3((n4 + 255) / 256), 256, 0, stream>>>(s, d, n4);
  };
  cast(x, x_h, (size_t)M * EMB);
  cast(Wq, Wq_h, (size_t)EMB * EMB);
  cast(WaK, WaK_h, (size_t)EMB * LAT);
  cast(WbK, WbK_h, (size_t)LAT * EMB);
  cast(WaV, WaV_h, (size_t)EMB * LAT);
  cast(WbV, WbV_h, (size_t)LAT * EMB);
  cast(Wo, Wo_h, (size_t)EMB * EMB);
  cast(W1, W1_h, (size_t)EMB * FFN);
  cast(W2, W2_h, (size_t)FFN * EMB);

  // Q / K / V projections
  gemm_kernel<false, _Float16><<<dim3(EMB / 128, M / 128), 256, 0, stream>>>(
      x_h, Wq_h, bq, q_h, M, EMB, EMB);
  gemm_kernel<false, _Float16><<<dim3(LAT / 128, M / 128), 256, 0, stream>>>(
      x_h, WaK_h, baK, ka_h, M, EMB, LAT);
  gemm_kernel<false, _Float16><<<dim3(EMB / 128, M / 128), 256, 0, stream>>>(
      ka_h, WbK_h, bbK, k_h, M, LAT, EMB);
  gemm_kernel<false, _Float16><<<dim3(LAT / 128, M / 128), 256, 0, stream>>>(
      x_h, WaV_h, baV, va_h, M, EMB, LAT);
  gemm_kernel<false, _Float16><<<dim3(EMB / 128, M / 128), 256, 0, stream>>>(
      va_h, WbV_h, bbV, v_h, M, LAT, EMB);

  // attention: grid (B*H, N/64), 4 waves per block
  attn_kernel<<<dim3(32, Nseq / 64), 128, 0, stream>>>(q_h, k_h, v_h, abias, attn_h);

  // output projection + residual LN1
  gemm_kernel<false, float><<<dim3(EMB / 128, M / 128), 256, 0, stream>>>(
      attn_h, Wo_h, bo, attnO, M, EMB, EMB);
  ln_kernel<true, true><<<dim3(M), 256, 0, stream>>>(x, attnO, ln1w, ln1b, h32, h16);

  // FFN (+exact GELU) and residual LN2 -> d_out
  gemm_kernel<true, _Float16><<<dim3(FFN / 128, M / 128), 256, 0, stream>>>(
      h16, W1_h, b1, ffn1_h, M, EMB, FFN);
  gemm_kernel<false, float><<<dim3(EMB / 128, M / 128), 256, 0, stream>>>(
      ffn1_h, W2_h, b2, ffn2, M, FFN, EMB);
  ln_kernel<true, false><<<dim3(M), 256, 0, stream>>>(h32, ffn2, ln2w, ln2b,
                                                      (float*)d_out, nullptr);
}
